// BruxismFrameJudgementLayer_52166672778108
// MI455X (gfx1250) — compile-verified, matrix operand order checked by reading, not run on hardware
//
#include <hip/hip_runtime.h>
#include <hip/hip_bf16.h>

typedef __attribute__((ext_vector_type(16))) _Float16 v16h;
typedef __attribute__((ext_vector_type(8)))  float    v8f;

// Exact rank count over all 512 scores for threshold t = score[jidx]:
//   rank = #{ i : s[i] > t  ||  (s[i] == t && i < jidx) }
// Implemented as a 16x32 f16 indicator matrix (the entire 512-vector, one
// element per (lane, half) slot) multiplied by an all-ones B via
// v_wmma_f32_16x16x32_f16. D[m][n] = rowsum(m) for every column n, so the
// full count is (sum of lane0's 8 accumulators) + (sum of lane16's 8
// accumulators): rows 0-7 and 8-15 respectively. Counts <= 512 are exact in
// f32; indicators 0.0/1.0 are exact in f16.
__device__ __forceinline__ float wave_rank_count(const float v[16], int base, float t, int jidx) {
    v16h a;
#pragma unroll
    for (int h = 0; h < 16; ++h) {
        const bool ind = (v[h] > t) || ((v[h] == t) && ((base + h) < jidx));
        a[h] = ind ? (_Float16)1.0f : (_Float16)0.0f;
    }
    v16h ones;
#pragma unroll
    for (int h = 0; h < 16; ++h) ones[h] = (_Float16)1.0f;

    v8f c = {};
    // 8 args: (neg_a, A, neg_b, B, c_mod, C, reuse_a, reuse_b)
    c = __builtin_amdgcn_wmma_f32_16x16x32_f16(
        /*neg_a=*/false, a, /*neg_b=*/false, ones,
        /*c_mod=*/(short)0, c, /*reuse_a=*/false, /*reuse_b=*/false);

    const float local = ((c[0] + c[1]) + (c[2] + c[3])) + ((c[4] + c[5]) + (c[6] + c[7]));
    const int li = __float_as_int(local);
    const float lo = __int_as_float(__builtin_amdgcn_readlane(li, 0));   // rows 0..7
    const float hi = __int_as_float(__builtin_amdgcn_readlane(li, 16));  // rows 8..15
    return lo + hi;
}

__global__ void __launch_bounds__(32)
brux_judgement_kernel(const float* __restrict__ score, float* __restrict__ out) {
    const int lane = threadIdx.x;            // one wave32, lanes 0..31
    const int base = lane * 16;              // this lane owns scores [base, base+16)

    // 64B per lane via four b128 loads.
    float v[16];
    const float4* s4 = (const float4*)score;
#pragma unroll
    for (int q = 0; q < 4; ++q) {
        const float4 x = s4[lane * 4 + q];
        v[q * 4 + 0] = x.x; v[q * 4 + 1] = x.y;
        v[q * 4 + 2] = x.z; v[q * 4 + 3] = x.w;
    }

    // Non-bruxism thresholds (uniform scalar loads).
    const float t38 = score[38];
    const float t42 = score[42];
    const float t0  = score[0];

    // Three WMMA-based rank counts; EXEC is all-1s here (no divergence yet).
    const float r38 = wave_rank_count(v, base, t38, 38);
    const float r42 = wave_rank_count(v, base, t42, 42);
    const float r0  = wave_rank_count(v, base, t0,   0);
    const bool has_non_brux = (r38 < 10.0f) || (r42 < 10.0f) || (r0 < 10.0f);

    if (lane == 0) {
        const int   cls[13][3] = {
            {500, 41,   0}, {369, 500,   0}, {412, 431, 470}, {127, 67, 103},
            {399, 403, 412}, {410, 411, 398}, {412, 410, 398}, {435, 438, 449},
            {374, 436, 435}, {372, 434, 469}, {500, 439,  50}, {399, 403, 410},
            {410, 398, 435}};
        const float mn[13][3] = {
            {0.1f, 0.05f, 0.0f}, {0.1f, 0.1f, 0.0f}, {0.05f, 0.05f, 0.05f},
            {0.1f, 0.1f, 0.1f}, {0.1f, 0.1f, 0.1f}, {0.1f, 0.1f, 0.1f},
            {0.1f, 0.1f, 0.1f}, {0.1f, 0.1f, 0.1f}, {0.1f, 0.1f, 0.1f},
            {0.1f, 0.1f, 0.1f}, {0.1f, 0.1f, 0.1f}, {0.1f, 0.1f, 0.1f},
            {0.1f, 0.1f, 0.1f}};
        const float mx[13][3] = {
            {0.8f, 0.5f, 1.0f}, {0.5f, 0.5f, 1.0f}, {0.4f, 0.4f, 0.4f},
            {0.8f, 0.7f, 0.6f}, {0.9f, 0.8f, 0.7f}, {0.9f, 0.8f, 0.7f},
            {0.8f, 0.7f, 0.6f}, {0.9f, 0.8f, 0.7f}, {0.9f, 0.8f, 0.7f},
            {0.9f, 0.8f, 0.7f}, {0.8f, 0.7f, 0.6f}, {0.9f, 0.8f, 0.7f},
            {0.9f, 0.8f, 0.7f}};
        const float brux[13] = {2.94f, 1.76f, 1.76f, 1.47f, 0.29f, 0.88f, 0.01f,
                                0.29f, 0.29f, 0.01f, 0.29f, 0.01f, 0.01f};

        int first = -1;
#pragma unroll
        for (int cidx = 0; cidx < 13; ++cidx) {
            bool pass = true;
#pragma unroll
            for (int k = 0; k < 3; ++k) {
                const float sv = score[cls[cidx][k]];
                pass = pass && (sv >= mn[cidx][k]) && (sv <= mx[cidx][k]);
            }
            if (pass && first < 0) first = cidx;
        }

        const bool judgement = (first >= 0) && !has_non_brux;
        const int fi = (first >= 0) ? first : 0;
        // jnp.round is round-half-to-even -> rintf (default RNE on device).
        const float value = judgement ? (rintf(brux[fi] * 100.0f) / 100.0f) : 0.0f;

        out[0] = judgement ? 1.0f : 0.0f;
        out[1] = value;
    }
}

extern "C" void kernel_launch(void* const* d_in, const int* in_sizes, int n_in,
                              void* d_out, int out_size, void* d_ws, size_t ws_size,
                              hipStream_t stream) {
    (void)in_sizes; (void)n_in; (void)out_size; (void)d_ws; (void)ws_size;
    const float* score = (const float*)d_in[0];   // (1, 512) f32
    float* out = (float*)d_out;                   // [judgement, value]
    brux_judgement_kernel<<<1, 32, 0, stream>>>(score, out);
}
